// Attention_10222022164769
// MI455X (gfx1250) — matrix operand order repaired and checked by reading, not compiled, not run on hardware
//
#include <hip/hip_runtime.h>
#include <hip/hip_bf16.h>
#include <cstdint>

// ---------------------------------------------------------------------------
// CDNA5 (gfx1250) fused attention: QKV GEMM + RoPE -> flash attention -> proj
// v_wmma_f32_16x16x32_bf16 compute; ALL tile movement in the GEMM loops rides
// the async-copy path (global_load_async_to_lds_b128 / ASYNCcnt), so barrier
// waitcnt flushes (LOADcnt) are no-ops in steady state.
// ---------------------------------------------------------------------------

typedef __bf16 bf16_t;
typedef __attribute__((ext_vector_type(16))) __bf16 v16bf;
typedef __attribute__((ext_vector_type(8)))  __bf16 v8bf;
typedef __attribute__((ext_vector_type(4)))  __bf16 v4bf;
typedef __attribute__((ext_vector_type(8)))  float  v8f;

#define BATCH 2
#define SEQ   2048
#define CDIM  1024
#define HEADS 16
#define HDIM  64
#define MROWS (BATCH * SEQ)   // 4096
#define DQKV  (3 * CDIM)      // 3072

// A fragment: 16x32 bf16 (M x K). lane<16: row=lane, K chunks 0..7 & 16..23;
// lane>=16: row=lane-16, K chunks 8..15 & 24..31. (ISA 7.12.2)
union AF { v16bf v; v8bf h[2]; };
// B fragment: 32x16 bf16 (K x N). lane<16: col=lane, K chunks 0..7 & 16..23;
// lane>=16: col=lane-16, K chunks 8..15 & 24..31.
union BF { v16bf v; v8bf h[2]; };

__device__ __forceinline__ v8f v8f_zero() {
  v8f z = {0.f, 0.f, 0.f, 0.f, 0.f, 0.f, 0.f, 0.f};
  return z;
}

__device__ __forceinline__ v8f wmma_bf16(const AF& a, const BF& b, v8f c) {
  return __builtin_amdgcn_wmma_f32_16x16x32_bf16(
      false, a.v, false, b.v, (short)0, c, false, false);
}

// one 16B async copy global -> LDS (per-lane addresses), tracked by ASYNCcnt.
// No "memory" clobber: ordering vs consumers is provided by __syncthreads();
// volatile keeps the instruction pinned relative to barriers / other asm.
__device__ __forceinline__ void async_b128(unsigned lds_off, unsigned voff,
                                           const void* sbase) {
  asm volatile("global_load_async_to_lds_b128 %0, %1, %2"
               :: "v"(lds_off), "v"(voff), "s"(sbase));
}
__device__ __forceinline__ void wait_async_le6() {
  asm volatile("s_wait_asynccnt 0x6");
}
__device__ __forceinline__ void wait_async_le4() {
  asm volatile("s_wait_asynccnt 0x4");
}
__device__ __forceinline__ void wait_async_le0() {
  asm volatile("s_wait_asynccnt 0x0");
}
__device__ __forceinline__ void wait_ds0() {
  asm volatile("s_wait_dscnt 0x0" ::: "memory");
}
__device__ __forceinline__ unsigned lds_addr_of(const void* p) {
  return (unsigned)(uintptr_t)p;   // low 32 bits of LDS aperture = byte offset
}

// reductions across the 16-lane half-groups (rows of the C-layout live there)
__device__ __forceinline__ float half16_max(float v) {
  v = fmaxf(v, __shfl_xor(v, 1));
  v = fmaxf(v, __shfl_xor(v, 2));
  v = fmaxf(v, __shfl_xor(v, 4));
  v = fmaxf(v, __shfl_xor(v, 8));
  return v;
}
__device__ __forceinline__ float half16_sum(float v) {
  v += __shfl_xor(v, 1);
  v += __shfl_xor(v, 2);
  v += __shfl_xor(v, 4);
  v += __shfl_xor(v, 8);
  return v;
}

// ---------------------------------------------------------------------------
// Block GEMM: 128(M) x 64(N), K = CDIM. Per k-step the block async-stages the
// 128x32 A chunk (8 KB) AND the 64x32 W chunk (4 KB) into double-buffered LDS
// (6 async b128 per thread), then reads WMMA fragments via ds_load_b128.
// ---------------------------------------------------------------------------
__device__ __forceinline__ void gemm_block_128x64(
    const bf16_t* __restrict__ abase,   // &A[m0_block * CDIM]
    const bf16_t* __restrict__ wbase,   // &W[d0 * CDIM]
    int tid, v8f acc[2][4])
{
  __shared__ bf16_t ldsA[2][4096];      // 2 x 8 KB : 128 rows x 32 bf16
  __shared__ bf16_t ldsB[2][2048];      // 2 x 4 KB :  64 rows x 32 bf16

  const int lane = tid & 31;
  const int wave = tid >> 5;
  const int half = lane >> 4;
  const int r    = lane & 15;

  const unsigned lA   = lds_addr_of(&ldsA[0][0]);
  const unsigned lB   = lds_addr_of(&ldsB[0][0]);
  const unsigned my16 = (unsigned)tid * 16u;
  // chunk c covers source row (c>>2), 16B piece (c&3); c+128 -> +32 rows
  const unsigned voff = (unsigned)((tid >> 2) * (CDIM * 2) + (tid & 3) * 16);

  auto stage = [&](int k, int s) {
    const bf16_t* ga = abase + k;
    const bf16_t* gw = wbase + k;
    const unsigned sA = lA + (unsigned)s * 8192u + my16;
    const unsigned sB = lB + (unsigned)s * 4096u + my16;
#pragma unroll
    for (int i = 0; i < 4; ++i)       // 4 x 32 rows of A
      async_b128(sA + (unsigned)i * 2048u, voff + (unsigned)i * 65536u, ga);
#pragma unroll
    for (int i = 0; i < 2; ++i)       // 2 x 32 rows of W
      async_b128(sB + (unsigned)i * 2048u, voff + (unsigned)i * 65536u, gw);
  };

  stage(0, 0);
#pragma unroll 2
  for (int k = 0; k < CDIM; k += 32) {
    const int cur = (k >> 5) & 1;
    const int nxt = cur ^ 1;
    if (k + 32 < CDIM) {
      stage(k + 32, nxt);
      wait_async_le6();               // cur's 6 copies done; nxt's in flight
    } else {
      wait_async_le0();
    }
    __syncthreads();                  // buf[cur] visible to all waves

    const bf16_t* la = &ldsA[cur][0] + (wave * 32 + r) * 32 + half * 8;
    const bf16_t* lb = &ldsB[cur][0] + r * 32 + half * 8;
    AF a[2];
    BF b[4];
#pragma unroll
    for (int i = 0; i < 2; ++i) {     // +16 rows = 512 elements
      a[i].h[0] = *(const v8bf*)(la + i * 512);
      a[i].h[1] = *(const v8bf*)(la + i * 512 + 16);
    }
#pragma unroll
    for (int t = 0; t < 4; ++t) {
      b[t].h[0] = *(const v8bf*)(lb + t * 512);
      b[t].h[1] = *(const v8bf*)(lb + t * 512 + 16);
    }
    __builtin_amdgcn_sched_barrier(0);  // keep all ds reads grouped
    wait_ds0();
    __syncthreads();                  // buf[cur] free for next prefetch
    __builtin_amdgcn_sched_barrier(0);

#pragma unroll
    for (int t = 0; t < 4; ++t) {
      acc[0][t] = wmma_bf16(a[0], b[t], acc[0][t]);
      acc[1][t] = wmma_bf16(a[1], b[t], acc[1][t]);
    }
  }
}

// ---------------------------------------------------------------------------
// Kernel 1: f32 -> bf16 conversion (vectorized x4)
// ---------------------------------------------------------------------------
__global__ __launch_bounds__(256) void cvt_f32_bf16(
    const float4* __restrict__ src, v4bf* __restrict__ dst, int n4) {
  int i = blockIdx.x * blockDim.x + threadIdx.x;
  if (i < n4) {
    float4 f = src[i];
    v4bf o;
    o[0] = (bf16_t)f.x; o[1] = (bf16_t)f.y;
    o[2] = (bf16_t)f.z; o[3] = (bf16_t)f.w;
    dst[i] = o;
  }
}

// ---------------------------------------------------------------------------
// Kernel 2: QKV GEMM + RoPE. Block: 128(M) x 64(D). Grid: (3072/64, 4096/128)
// ---------------------------------------------------------------------------
__global__ __launch_bounds__(128) void qkv_rope_kernel(
    const bf16_t* __restrict__ xb,     // [4096, 1024]
    const bf16_t* __restrict__ wqkv,   // [3072, 1024]
    const float*  __restrict__ cosb,   // [2048, 64]
    const float*  __restrict__ sinb,   // [2048, 64]
    bf16_t* __restrict__ qb,           // [B*H, 2048, 64]  (pre-scaled by 1/8)
    bf16_t* __restrict__ kb,           // [B*H, 2048, 64]
    bf16_t* __restrict__ vT)           // [B*H, 64, 2048]  (transposed)
{
  const int tid  = threadIdx.x;
  const int lane = tid & 31;
  const int wave = tid >> 5;
  const int half = lane >> 4;
  const int r    = lane & 15;
  const int d0   = blockIdx.x * 64;
  const int m0   = blockIdx.y * 128 + wave * 32;

  v8f acc[2][4];
#pragma unroll
  for (int i = 0; i < 2; ++i)
#pragma unroll
    for (int t = 0; t < 4; ++t) acc[i][t] = v8f_zero();

  gemm_block_128x64(xb + (size_t)blockIdx.y * 128 * CDIM,
                    wqkv + (size_t)d0 * CDIM, tid, acc);

  const int which = d0 >> 10;         // 0=q, 1=k, 2=v
  const int h     = (d0 & 1023) >> 6; // head index

#pragma unroll
  for (int i = 0; i < 2; ++i) {
#pragma unroll
    for (int j = 0; j < 8; ++j) {
      const int m = m0 + i * 16 + j + 8 * half;
      const int bidx = m >> 11;
      const int n    = m & 2047;
      const int bh   = bidx * HEADS + h;

      if (which == 2) {
#pragma unroll
        for (int t = 0; t < 4; ++t) {
          const int hd = t * 16 + r;
          vT[((size_t)bh * HDIM + hd) * SEQ + n] = (bf16_t)acc[i][t][j];
        }
      } else {
        bf16_t* dst = (which == 0) ? qb : kb;
        const float sc = (which == 0) ? 0.125f : 1.0f;
        const size_t base = ((size_t)bh * SEQ + n) * HDIM;
#pragma unroll
        for (int t = 0; t < 2; ++t) {
          const int hd1 = t * 16 + r;
          const int hd2 = hd1 + 32;
          const float c1 = cosb[n * HDIM + hd1], s1 = sinb[n * HDIM + hd1];
          const float c2 = cosb[n * HDIM + hd2], s2 = sinb[n * HDIM + hd2];
          const float x1 = acc[i][t][j];
          const float x2 = acc[i][t + 2][j];
          dst[base + hd1] = (bf16_t)((x1 * c1 - x2 * s1) * sc);
          dst[base + hd2] = (bf16_t)((x2 * c2 + x1 * s2) * sc);
        }
      }
    }
  }
}

// ---------------------------------------------------------------------------
// Kernel 3: flash attention. One (b,h) per blockIdx.x, 16 q rows per wave.
// K (32x64) and V (64x32) blocks async-staged to LDS once per block per step,
// double buffered; softmax VALU overlaps the next stage's copies.
// ---------------------------------------------------------------------------
__global__ __launch_bounds__(128) void flash_kernel(
    const bf16_t* __restrict__ qb,
    const bf16_t* __restrict__ kb,
    const bf16_t* __restrict__ vT,
    bf16_t* __restrict__ attn_out)     // [B, N, H*HD]
{
  __shared__ bf16_t ldsK[2][2048];     // 2 x 4 KB: 32 rows x 64 bf16
  __shared__ bf16_t ldsV[2][2048];     // 2 x 4 KB: 64 rows x 32 bf16
  __shared__ bf16_t ldsP[4][16 * 40];  // per-wave P tile, 80B row stride

  const int tid  = threadIdx.x;
  const int lane = tid & 31;
  const int wave = tid >> 5;
  const int half = lane >> 4;
  const int r    = lane & 15;
  const int bh   = blockIdx.x;
  const int bidx = bh >> 4;
  const int h    = bh & 15;
  const int q0   = blockIdx.y * 64 + wave * 16;

  const bf16_t* Q = qb + (size_t)bh * SEQ * HDIM;
  const bf16_t* K = kb + (size_t)bh * SEQ * HDIM;
  const bf16_t* V = vT + (size_t)bh * HDIM * SEQ;   // [64, 2048]

  const unsigned ldsK0 = lds_addr_of(&ldsK[0][0]);
  const unsigned ldsV0 = lds_addr_of(&ldsV[0][0]);
  const unsigned myoff = (unsigned)tid * 16u;
  // K tile: chunk c covers row c>>3 (128B rows), piece c&7; c+128 -> +16 rows
  const unsigned voffK = (unsigned)((tid >> 3) * (HDIM * 2) + (tid & 7) * 16);
  // V tile: chunk c covers vT row c>>2 (64B pieces of 4096B rows); +32 rows
  const unsigned voffV = (unsigned)((tid >> 2) * (SEQ * 2) + (tid & 3) * 16);

  auto stage = [&](int jb, int s) {
    const bf16_t* gk = K + (size_t)jb * HDIM;
    const bf16_t* gv = V + jb;
    const unsigned so = (unsigned)s * 4096u;
    async_b128(ldsK0 + so + myoff,         voffK,            gk);
    async_b128(ldsK0 + so + 2048u + myoff, voffK + 2048u,    gk);
    async_b128(ldsV0 + so + myoff,         voffV,            gv);
    async_b128(ldsV0 + so + 2048u + myoff, voffV + 131072u,  gv);
  };

  // Q fragments are loop-invariant
  AF qa[2];
  {
    const bf16_t* qrow = Q + (size_t)(q0 + r) * HDIM + half * 8;
    qa[0].h[0] = *(const v8bf*)(qrow);
    qa[0].h[1] = *(const v8bf*)(qrow + 16);
    qa[1].h[0] = *(const v8bf*)(qrow + 32);
    qa[1].h[1] = *(const v8bf*)(qrow + 48);
  }

  v8f o[4] = {v8f_zero(), v8f_zero(), v8f_zero(), v8f_zero()};
  float mrow[8], lrow[8];
#pragma unroll
  for (int j = 0; j < 8; ++j) { mrow[j] = -3.0e38f; lrow[j] = 0.f; }

  bf16_t* myP = ldsP[wave];
  stage(0, 0);

#pragma unroll 2
  for (int jb = 0; jb < SEQ; jb += 32) {
    const int cur = (jb >> 5) & 1;
    const int nxt = cur ^ 1;
    if (jb + 32 < SEQ) {
      stage(jb + 32, nxt);
      wait_async_le4();               // cur's 4 copies done; nxt's in flight
    } else {
      wait_async_le0();
    }
    __syncthreads();                  // K/V buf[cur] visible to all waves

    // fragment reads from LDS (grouped so they pipeline under one wait)
    BF bk[4], bv[4];
    const bf16_t* lk = &ldsK[cur][0];
    const bf16_t* lv = &ldsV[cur][0];
#pragma unroll
    for (int d = 0; d < 2; ++d)
#pragma unroll
      for (int tile = 0; tile < 2; ++tile) {
        const bf16_t* kp = lk + (tile * 16 + r) * 64 + d * 32 + half * 8;
        bk[d * 2 + tile].h[0] = *(const v8bf*)(kp);
        bk[d * 2 + tile].h[1] = *(const v8bf*)(kp + 16);
      }
#pragma unroll
    for (int t = 0; t < 4; ++t) {
      const bf16_t* vp = lv + (t * 16 + r) * 32 + half * 8;
      bv[t].h[0] = *(const v8bf*)(vp);
      bv[t].h[1] = *(const v8bf*)(vp + 16);
    }
    __builtin_amdgcn_sched_barrier(0);
    wait_ds0();
    __syncthreads();                  // buf[cur] free; compute overlaps staging
    __builtin_amdgcn_sched_barrier(0);

    // ---- S = Q * K^T for 32 keys ----
    v8f s0 = v8f_zero(), s1 = v8f_zero();
#pragma unroll
    for (int d = 0; d < 2; ++d) {
      s0 = wmma_bf16(qa[d], bk[d * 2 + 0], s0);
      s1 = wmma_bf16(qa[d], bk[d * 2 + 1], s1);
    }

    // ---- online softmax update ----
#pragma unroll
    for (int j = 0; j < 8; ++j) {
      float mx = half16_max(fmaxf(s0[j], s1[j]));
      const float mnew = fmaxf(mrow[j], mx);
      const float corr = __expf(mrow[j] - mnew);
      const float p0 = __expf(s0[j] - mnew);
      const float p1 = __expf(s1[j] - mnew);
      lrow[j] = lrow[j] * corr + half16_sum(p0 + p1);
      mrow[j] = mnew;
#pragma unroll
      for (int t = 0; t < 4; ++t) o[t][j] *= corr;
      const int row = j + 8 * half;
      myP[row * 40 + r]      = (bf16_t)p0;
      myP[row * 40 + 16 + r] = (bf16_t)p1;
    }

    // wave-private LDS round trip: C-layout -> A-fragment layout
    wait_ds0();
    AF pa;
    {
      const bf16_t* prow = myP + r * 40 + half * 8;
      pa.h[0] = *(const v8bf*)(prow);
      pa.h[1] = *(const v8bf*)(prow + 16);
    }

    // ---- O += P * V ----
#pragma unroll
    for (int t = 0; t < 4; ++t) o[t] = wmma_bf16(pa, bv[t], o[t]);
  }

#pragma unroll
  for (int j = 0; j < 8; ++j) {
    const float inv = 1.0f / lrow[j];
    const int n = q0 + j + 8 * half;
    const size_t base = ((size_t)bidx * SEQ + n) * CDIM + h * HDIM;
#pragma unroll
    for (int t = 0; t < 4; ++t)
      attn_out[base + t * 16 + r] = (bf16_t)(o[t][j] * inv);
  }
}

// ---------------------------------------------------------------------------
// Kernel 4: output projection + bias (f32 out). Block: 128(M) x 64(D).
// ---------------------------------------------------------------------------
__global__ __launch_bounds__(128) void proj_kernel(
    const bf16_t* __restrict__ ab,     // [4096, 1024]
    const bf16_t* __restrict__ wp,     // [1024, 1024]
    const float*  __restrict__ bias,   // [1024]
    float* __restrict__ out)           // [4096, 1024]
{
  const int tid  = threadIdx.x;
  const int lane = tid & 31;
  const int wave = tid >> 5;
  const int half = lane >> 4;
  const int r    = lane & 15;
  const int d0   = blockIdx.x * 64;
  const int m0   = blockIdx.y * 128 + wave * 32;

  v8f acc[2][4];
#pragma unroll
  for (int i = 0; i < 2; ++i)
#pragma unroll
    for (int t = 0; t < 4; ++t) acc[i][t] = v8f_zero();

  gemm_block_128x64(ab + (size_t)blockIdx.y * 128 * CDIM,
                    wp + (size_t)d0 * CDIM, tid, acc);

#pragma unroll
  for (int i = 0; i < 2; ++i) {
#pragma unroll
    for (int j = 0; j < 8; ++j) {
      const int m = m0 + i * 16 + j + 8 * half;
#pragma unroll
      for (int t = 0; t < 4; ++t) {
        const int d = d0 + t * 16 + r;
        out[(size_t)m * CDIM + d] = acc[i][t][j] + bias[d];
      }
    }
  }
}

// ---------------------------------------------------------------------------
// Launch: converts -> qkv+rope -> flash -> proj (stream-ordered)
// ---------------------------------------------------------------------------
extern "C" void kernel_launch(void* const* d_in, const int* in_sizes, int n_in,
                              void* d_out, int out_size, void* d_ws, size_t ws_size,
                              hipStream_t stream) {
  const float* x      = (const float*)d_in[0];
  const float* cosb   = (const float*)d_in[1];
  const float* sinb   = (const float*)d_in[2];
  const float* qkv_w  = (const float*)d_in[3];
  const float* proj_w = (const float*)d_in[4];
  const float* proj_b = (const float*)d_in[5];
  float* out = (float*)d_out;
  char* ws = (char*)d_ws;

  bf16_t* xb     = (bf16_t*)(ws);                    //  8 MB
  bf16_t* wqkvb  = (bf16_t*)(ws + 8388608);          //  6 MB
  bf16_t* wprojb = (bf16_t*)(ws + 14680064);         //  2 MB
  bf16_t* qb     = (bf16_t*)(ws + 16777216);         //  8 MB
  bf16_t* kb     = (bf16_t*)(ws + 25165824);         //  8 MB
  bf16_t* vT     = (bf16_t*)(ws + 33554432);         //  8 MB
  bf16_t* ab     = (bf16_t*)(ws + 41943040);         //  8 MB -> 48 MB total

  cvt_f32_bf16<<<4096, 256, 0, stream>>>((const float4*)x,      (v4bf*)xb,     1048576);
  cvt_f32_bf16<<<3072, 256, 0, stream>>>((const float4*)qkv_w,  (v4bf*)wqkvb,   786432);
  cvt_f32_bf16<<<1024, 256, 0, stream>>>((const float4*)proj_w, (v4bf*)wprojb,  262144);

  qkv_rope_kernel<<<dim3(DQKV / 64, MROWS / 128), 128, 0, stream>>>(
      xb, wqkvb, cosb, sinb, qb, kb, vT);

  flash_kernel<<<dim3(BATCH * HEADS, SEQ / 64), 128, 0, stream>>>(
      qb, kb, vT, ab);

  proj_kernel<<<dim3(CDIM / 64, MROWS / 128), 128, 0, stream>>>(
      ab, wprojb, proj_b, out);
}
